// VQVAE_22308060135448
// MI455X (gfx1250) — matrix operand order repaired and checked by reading, not compiled, not run on hardware
//
#include <hip/hip_runtime.h>
#include <stddef.h>

#define D     256
#define KCB   8192
#define NROW  16384
#define DP    264   // padded LDS row stride (264*2B = 528B -> 4-bank rotation per row)

typedef __bf16 bf16_t;
typedef __attribute__((ext_vector_type(16))) __bf16 v16bf;
typedef __attribute__((ext_vector_type(8)))  float  v8f;

// ---------------------------------------------------------------------------
// K1: codebook fp32 -> bf16 hi/lo split + per-row squared norms.
// One wave per codebook row; lane d-stride 32 keeps loads coalesced.
// ---------------------------------------------------------------------------
__global__ __launch_bounds__(256) void vq_prep(const float* __restrict__ cb,
                                               bf16_t* __restrict__ cb_hi,
                                               bf16_t* __restrict__ cb_lo,
                                               float* __restrict__ cbnorm)
{
    const int wave = threadIdx.x >> 5;
    const int lane = threadIdx.x & 31;
    const int row  = blockIdx.x * 8 + wave;
    const float* src = cb + (size_t)row * D;
    float s = 0.0f;
#pragma unroll
    for (int j = 0; j < 8; ++j) {
        const int d = lane + 32 * j;
        const float x = src[d];
        const bf16_t h = (bf16_t)x;
        const bf16_t l = (bf16_t)(x - (float)h);
        cb_hi[(size_t)row * D + d] = h;
        cb_lo[(size_t)row * D + d] = l;
        s += x * x;
    }
#pragma unroll
    for (int mk = 16; mk >= 1; mk >>= 1) s += __shfl_xor(s, mk, 32);
    if (lane == 0) cbnorm[row] = s;
}

// ---------------------------------------------------------------------------
// K2: fused distance-GEMM + argmin.
// Block = 16 z-rows (M-tile), 8 waves split K=8192 into 64 16-entry tiles each.
// A (z) fragments: built once from LDS in this code's contiguous k order:
//   lane l (m = l&15, koff = (l>>4)*16): elems 0..15 -> k=koff..koff+15
//   (contiguous, matching the B fragment k order).
// B (codebook^T) fragments: same contiguous k order as A => lane l holds column
//   n = l&15, k = (l>>4)*16 + e  -> 16 CONTIGUOUS bf16 per lane: direct
//   32B-aligned global load, no transpose needed.
// C layout: lane l, vgpr r -> M = r + 8*(l>>4), N = l&15.
// ---------------------------------------------------------------------------
__global__ __launch_bounds__(256) void vq_argmin(const float* __restrict__ z,
                                                 const bf16_t* __restrict__ cb_hi,
                                                 const bf16_t* __restrict__ cb_lo,
                                                 const float* __restrict__ cbnorm,
                                                 int* __restrict__ widx)
{
    __shared__ bf16_t lds_hi[16][DP];
    __shared__ bf16_t lds_lo[16][DP];
    __shared__ float  red_s[8][16];
    __shared__ int    red_i[8][16];

    const int tid  = threadIdx.x;
    const int wave = tid >> 5;
    const int lane = tid & 31;
    const int m0   = blockIdx.x * 16;

    // Stage 1: load 16x256 z tile, split into bf16 hi/lo in LDS (coalesced).
#pragma unroll
    for (int r = 0; r < 16; ++r) {
        const float x = z[(size_t)(m0 + r) * D + tid];
        const bf16_t h = (bf16_t)x;
        lds_hi[r][tid] = h;
        lds_lo[r][tid] = (bf16_t)(x - (float)h);
    }
    __syncthreads();

    // Stage 2: persistent A fragments (8 D-chunks), reused for all 64 K-tiles.
    const int mrow  = lane & 15;
    const int koffA = (lane >> 4) * 16;
    v16bf a_hi[8], a_lo[8];
#pragma unroll
    for (int c = 0; c < 8; ++c) {
        const int d0 = 32 * c + koffA;
#pragma unroll
        for (int e = 0; e < 8; ++e) {
            a_hi[c][e]     = lds_hi[mrow][d0 + e];
            a_lo[c][e]     = lds_lo[mrow][d0 + e];
            a_hi[c][8 + e] = lds_hi[mrow][d0 + 8 + e];
            a_lo[c][8 + e] = lds_lo[mrow][d0 + 8 + e];
        }
    }

    const int ncol  = lane & 15;
    const int koffB = (lane >> 4) * 16;

    float best[8];
    int   bidx[8];
#pragma unroll
    for (int r = 0; r < 8; ++r) { best[r] = __builtin_inff(); bidx[r] = 0; }

    // Stage 3: scan codebook. Waves interleave tiles (e0 = (t*8+w)*16) so the
    // whole block walks the same L2-resident codebook region together.
    for (int t = 0; t < 64; ++t) {
        const int e0 = (t * 8 + wave) * 16;
        const int eidx = e0 + ncol;
        const bf16_t* bh_base = cb_hi + (size_t)eidx * D + koffB;
        const bf16_t* bl_base = cb_lo + (size_t)eidx * D + koffB;

        if (t + 1 < 64) {  // prefetch next tile (+128 codebook rows)
            __builtin_prefetch(bh_base + (size_t)128 * D, 0, 3);
            __builtin_prefetch(bl_base + (size_t)128 * D, 0, 3);
        }

        v8f acc = {};
#pragma unroll
        for (int c = 0; c < 8; ++c) {
            const v16bf bh = *(const v16bf*)(bh_base + 32 * c);
            const v16bf bl = *(const v16bf*)(bl_base + 32 * c);
            acc = __builtin_amdgcn_wmma_f32_16x16x32_bf16(false, a_hi[c], false, bh,
                                                          (short)0, acc, false, false);
            acc = __builtin_amdgcn_wmma_f32_16x16x32_bf16(false, a_hi[c], false, bl,
                                                          (short)0, acc, false, false);
            acc = __builtin_amdgcn_wmma_f32_16x16x32_bf16(false, a_lo[c], false, bh,
                                                          (short)0, acc, false, false);
        }

        const float en = cbnorm[eidx];
#pragma unroll
        for (int r = 0; r < 8; ++r) {
            const float s = en - 2.0f * acc[r];    // ||e||^2 - 2 z.e  (||z||^2 dropped)
            if (s < best[r]) { best[r] = s; bidx[r] = eidx; }  // strict < => first min
        }
    }

    // Stage 4: argmin across the 16 columns of each half-wave (xor shuffles).
#pragma unroll
    for (int r = 0; r < 8; ++r) {
#pragma unroll
        for (int mk = 8; mk >= 1; mk >>= 1) {
            const float os = __shfl_xor(best[r], mk, 16);
            const int   oi = __shfl_xor(bidx[r], mk, 16);
            if (os < best[r] || (os == best[r] && oi < bidx[r])) {
                best[r] = os; bidx[r] = oi;
            }
        }
    }
    if ((lane & 15) == 0) {
        const int mb = (lane >> 4) * 8;   // lane 0 -> rows 0..7, lane 16 -> rows 8..15
#pragma unroll
        for (int r = 0; r < 8; ++r) {
            red_s[wave][mb + r] = best[r];
            red_i[wave][mb + r] = bidx[r];
        }
    }
    __syncthreads();

    // Stage 5: argmin across the 8 waves (disjoint K-ranges).
    if (tid < 16) {
        float bs = red_s[0][tid];
        int   bi = red_i[0][tid];
#pragma unroll
        for (int w = 1; w < 8; ++w) {
            const float s = red_s[w][tid];
            const int   i = red_i[w][tid];
            if (s < bs || (s == bs && i < bi)) { bs = s; bi = i; }
        }
        widx[m0 + tid] = bi;
    }
}

// ---------------------------------------------------------------------------
// K3: z_e copy + codebook gather (z_q_st == z_q numerically) + per-row
// partial sum of (z_q - z_e)^2. No fp atomics -> bitwise deterministic.
// ---------------------------------------------------------------------------
__global__ __launch_bounds__(256) void vq_gather(const float* __restrict__ z,
                                                 const float* __restrict__ cb,
                                                 const int* __restrict__ widx,
                                                 float* __restrict__ out,
                                                 float* __restrict__ partial)
{
    const int row = blockIdx.x;
    const int d   = threadIdx.x;
    const int idx = widx[row];

    const float ze = z[(size_t)row * D + d];
    const float zq = cb[(size_t)idx * D + d];
    out[(size_t)row * D + d] = ze;                               // output 0: z_e
    out[(size_t)NROW * D + 1 + (size_t)row * D + d] = zq;        // output 2: z_q_st

    const float df = zq - ze;
    float s = df * df;
#pragma unroll
    for (int mk = 16; mk >= 1; mk >>= 1) s += __shfl_xor(s, mk, 32);

    __shared__ float red[8];
    if ((threadIdx.x & 31) == 0) red[threadIdx.x >> 5] = s;
    __syncthreads();
    if (threadIdx.x == 0) {
        float t = 0.0f;
#pragma unroll
        for (int i = 0; i < 8; ++i) t += red[i];
        partial[row] = t;
    }
}

// K4: final loss = (commitment + codebook) = 2 * mean((z_q - z_e)^2)
__global__ __launch_bounds__(256) void vq_loss(const float* __restrict__ partial,
                                               float* __restrict__ out)
{
    float s = 0.0f;
    for (int i = threadIdx.x; i < NROW; i += 256) s += partial[i];
#pragma unroll
    for (int mk = 16; mk >= 1; mk >>= 1) s += __shfl_xor(s, mk, 32);
    __shared__ float red[8];
    if ((threadIdx.x & 31) == 0) red[threadIdx.x >> 5] = s;
    __syncthreads();
    if (threadIdx.x == 0) {
        float t = 0.0f;
#pragma unroll
        for (int i = 0; i < 8; ++i) t += red[i];
        out[(size_t)NROW * D] = 2.0f * t / (float)(NROW * D);    // output 1: loss
    }
}

// ---------------------------------------------------------------------------
extern "C" void kernel_launch(void* const* d_in, const int* in_sizes, int n_in,
                              void* d_out, int out_size, void* d_ws, size_t ws_size,
                              hipStream_t stream)
{
    const float* z  = (const float*)d_in[0];   // [16384, 256]
    const float* cb = (const float*)d_in[1];   // [8192, 256]
    float* out = (float*)d_out;                // z_e | loss | z_q_st

    char* ws = (char*)d_ws;
    bf16_t* cb_hi   = (bf16_t*)(ws);                         // 4 MB
    bf16_t* cb_lo   = (bf16_t*)(ws + (size_t)4194304);       // 4 MB
    float*  cbnorm  = (float*) (ws + (size_t)8388608);       // 32 KB
    int*    widx    = (int*)   (ws + (size_t)8421376);       // 64 KB
    float*  partial = (float*) (ws + (size_t)8486912);       // 64 KB

    vq_prep  <<<KCB / 8,  256, 0, stream>>>(cb, cb_hi, cb_lo, cbnorm);
    vq_argmin<<<NROW / 16, 256, 0, stream>>>(z, cb_hi, cb_lo, cbnorm, widx);
    vq_gather<<<NROW,     256, 0, stream>>>(z, cb, widx, out, partial);
    vq_loss  <<<1,        256, 0, stream>>>(partial, out);
}
